// GPT_2989297238680
// MI455X (gfx1250) — matrix operand order repaired and checked by reading, not compiled, not run on hardware
//
#include <hip/hip_runtime.h>

typedef _Float16 half_t;
typedef __attribute__((ext_vector_type(16))) _Float16 v16h;
typedef __attribute__((ext_vector_type(8)))  _Float16 v8h;
typedef __attribute__((ext_vector_type(8)))  float    v8f;

#define WMMA_F16(A, B, C) \
  __builtin_amdgcn_wmma_f32_16x16x32_f16(false, (A), false, (B), (short)0, (C), false, false)

// Assemble a v16h fragment from two contiguous 8-half runs.
__device__ __forceinline__ v16h mk_frag(const half_t* p0, const half_t* p1) {
  v8h a = *(const v8h*)p0;
  v8h b = *(const v8h*)p1;
  v16h f;
#pragma unroll
  for (int i = 0; i < 8; ++i) { f[i] = a[i]; f[i + 8] = b[i]; }
  return f;
}

// ---------------------------------------------------------------------------
// Async global->LDS 16B copy: CDNA5 GLOBAL_LOAD_ASYNC_TO_LDS_B128 (ASYNCcnt).
// VDST = VGPR with LDS byte address (low 32 bits of generic shared pointer),
// VADDR = 64-bit global address, no SADDR. Inline asm avoids the clang
// builtin's CUDA-LangAS pointer types, which HIP source cannot spell.
// ---------------------------------------------------------------------------
__device__ __forceinline__ void async_copy16(const half_t* g, half_t* l) {
  unsigned int lds = (unsigned int)(size_t)l;
  asm volatile("global_load_async_to_lds_b128 %0, %1, off"
               :: "v"(lds), "v"(g) : "memory");
}

__device__ __forceinline__ void wait_async_lds() {
  asm volatile("s_wait_asynccnt 0" ::: "memory");
}

// ---------------------------------------------------------------------------
// Embedding: h[bt][e] = tok_emb[x[bt]][e] + pos_emb[t][e]   (f32 + f16 copies)
// ---------------------------------------------------------------------------
__global__ __launch_bounds__(256) void embed_kernel(
    const int* __restrict__ x, const float* __restrict__ tok,
    const float* __restrict__ pos, float* __restrict__ h,
    half_t* __restrict__ h16) {
  int bt = blockIdx.x;              // 0..2047
  int t  = bt & 1023;
  int tk = x[bt];
  size_t base = (size_t)bt * 1024;
  for (int i = threadIdx.x; i < 1024; i += 256) {
    float v = tok[(size_t)tk * 1024 + i] + pos[(size_t)t * 1024 + i];
    h[base + i]   = v;
    h16[base + i] = (half_t)v;
  }
}

// ---------------------------------------------------------------------------
// LayerNorm (row = 1024). Optional f32 and f16 outputs.
// ---------------------------------------------------------------------------
__global__ __launch_bounds__(256) void ln_kernel(
    const float* __restrict__ in, const float* __restrict__ g,
    const float* __restrict__ b, float* __restrict__ outF,
    half_t* __restrict__ outH) {
  __shared__ float r1[256], r2[256];
  int tid = threadIdx.x;
  size_t base = (size_t)blockIdx.x * 1024;
  float s = 0.f, s2 = 0.f;
  for (int i = tid; i < 1024; i += 256) {
    float v = in[base + i];
    s += v; s2 += v * v;
  }
  r1[tid] = s; r2[tid] = s2;
  __syncthreads();
  for (int off = 128; off > 0; off >>= 1) {
    if (tid < off) { r1[tid] += r1[tid + off]; r2[tid] += r2[tid + off]; }
    __syncthreads();
  }
  float mean = r1[0] * (1.f / 1024.f);
  float var  = r2[0] * (1.f / 1024.f) - mean * mean;
  float rs   = rsqrtf(var + 1e-5f);
  for (int i = tid; i < 1024; i += 256) {
    float y = (in[base + i] - mean) * rs * g[i] + b[i];
    if (outF) outF[base + i] = y;
    if (outH) outH[base + i] = (half_t)y;
  }
}

// ---------------------------------------------------------------------------
// Repack V: [b,t,h,hd] (f16) -> Vt [b,h,hd,t] (f16)
// ---------------------------------------------------------------------------
__global__ __launch_bounds__(256) void repackV_kernel(
    const half_t* __restrict__ v, half_t* __restrict__ vt) {
  int idx = blockIdx.x * 256 + threadIdx.x;   // < 2*1024*1024
  int bt = idx >> 10, e = idx & 1023;
  int b = bt >> 10, t = bt & 1023;
  int h = e >> 6, hd = e & 63;
  vt[(((size_t)(b * 16 + h) * 64 + hd) << 10) + t] = v[idx];
}

// ---------------------------------------------------------------------------
// B-tile fill: 32(k) x 128(n) f32 -> sB[n][k] f16 (transposed, stride 40)
// HEADB: B laid out as (H, E, HD): off = (n>>6)*65536 + k*64 + (n&63)
// ---------------------------------------------------------------------------
template <bool HEADB>
__device__ __forceinline__ void fill_b(half_t* __restrict__ dstB,
                                       const float* __restrict__ B, int k0,
                                       int bn, int N, int kr, int cs) {
  if (HEADB) {
#pragma unroll
    for (int i = 0; i < 16; ++i) {
      int n = bn + cs * 16 + i;
      float v = B[(size_t)(n >> 6) * 65536 + (size_t)(k0 + kr) * 64 + (n & 63)];
      dstB[(cs * 16 + i) * 40 + kr] = (half_t)v;
    }
  } else {
    const float* src = B + (size_t)(k0 + kr) * N + bn + cs * 16;
#pragma unroll
    for (int i = 0; i < 16; ++i)
      dstB[(cs * 16 + i) * 40 + kr] = (half_t)src[i];
  }
}

// ---------------------------------------------------------------------------
// WMMA GEMM: C[M,N] = A[M,K](f16) x B[K,N](f32, converted to f16 on LDS fill)
// Tile 128x128, BK=32, 256 threads = 8 waves, double-buffered LDS with async
// global->LDS copies for the A tile. One barrier per K-step.
// flags: 1=+bias  2=+residual  4=ReLU  8=write f32  16=write f16
// ---------------------------------------------------------------------------
template <bool HEADB>
__global__ __launch_bounds__(256) void gemm_f16_kernel(
    const half_t* __restrict__ A, const float* __restrict__ B,
    const float* __restrict__ bias, const float* __restrict__ resid,
    float* __restrict__ outF, half_t* __restrict__ outH,
    int N, int K, int flags) {
  __shared__ half_t sA[2][128 * 32];        // [m][k]
  __shared__ half_t sB[2][128 * 40];        // [n][k] transposed, padded

  const int tid  = threadIdx.x;
  const int lane = tid & 31;
  const int wave = tid >> 5;
  const int l15  = lane & 15;
  const int kh   = lane >> 4;
  const int bm   = blockIdx.y * 128;
  const int bn   = blockIdx.x * 128;

  const int arow = tid >> 1, aseg = tid & 1;   // A fill: 2 thr/row x 16 halfs
  const int bkr  = tid >> 3, bcs  = tid & 7;   // B fill: 16 cols each

  v8f acc[8];
#pragma unroll
  for (int i = 0; i < 8; ++i) acc[i] = (v8f){};

  // Prologue: fill buffer 0
  {
    const half_t* asrc = A + (size_t)(bm + arow) * K + aseg * 16;
    half_t* adst = &sA[0][arow * 32 + aseg * 16];
    async_copy16(asrc, adst);
    async_copy16(asrc + 8, adst + 8);
    fill_b<HEADB>(sB[0], B, 0, bn, N, bkr, bcs);
    if (!HEADB && 32 < K)
      __builtin_prefetch(B + (size_t)(32 + bkr) * N + bn + bcs * 16, 0, 1);
  }
  wait_async_lds();
  __syncthreads();

  int cur = 0;
  for (int k0 = 0; k0 < K; k0 += 32) {
    const int nk = k0 + 32;
    if (nk < K) {
      // Prefill next buffer (async A copy overlaps the WMMAs below)
      const half_t* asrc = A + (size_t)(bm + arow) * K + nk + aseg * 16;
      half_t* adst = &sA[cur ^ 1][arow * 32 + aseg * 16];
      async_copy16(asrc, adst);
      async_copy16(asrc + 8, adst + 8);
      fill_b<HEADB>(sB[cur ^ 1], B, nk, bn, N, bkr, bcs);
      if (!HEADB && nk + 32 < K)
        __builtin_prefetch(B + (size_t)(nk + 32 + bkr) * N + bn + bcs * 16, 0, 1);
    }

    // Compute on current buffer
    const half_t* ap = &sA[cur][(wave * 16 + l15) * 32 + kh * 8];
    v16h af = mk_frag(ap, ap + 16);
#pragma unroll
    for (int nf = 0; nf < 8; ++nf) {
      const half_t* bp = &sB[cur][(nf * 16 + l15) * 40 + kh * 8];
      v16h bf = mk_frag(bp, bp + 16);
      acc[nf] = WMMA_F16(af, bf, acc[nf]);
    }

    wait_async_lds();
    __syncthreads();
    cur ^= 1;
  }

  // Epilogue
#pragma unroll
  for (int nf = 0; nf < 8; ++nf) {
#pragma unroll
    for (int r = 0; r < 8; ++r) {
      int row = bm + wave * 16 + r + kh * 8;
      int col = bn + nf * 16 + l15;
      float v = acc[nf][r];
      if (flags & 1) v += bias[col];
      if (flags & 2) v += resid[(size_t)row * N + col];
      if (flags & 4) v = fmaxf(v, 0.f);
      size_t o = (size_t)row * N + col;
      if (flags & 8)  outF[o] = v;
      if (flags & 16) outH[o] = (half_t)v;
    }
  }
}

// ---------------------------------------------------------------------------
// Flash attention: 1 wave per (b, h, 16-query tile). Streams 32-key chunks.
// q,k: [b,t, h*64+hd] f16 ; vT: [b,h,hd,t] f16 ; o: [b,t, h*64+hd] f16
// ---------------------------------------------------------------------------
__global__ __launch_bounds__(32) void attn_kernel(
    const half_t* __restrict__ q, const half_t* __restrict__ k,
    const half_t* __restrict__ vT, half_t* __restrict__ o) {
  __shared__ half_t P[16 * 32];

  const int lane = threadIdx.x;
  const int l15  = lane & 15;
  const int kh   = lane >> 4;
  const int qt = blockIdx.x, hh = blockIdx.y, b = blockIdx.z;
  const int qbase = qt * 16;

  // Q fragments (K=64 -> two K=32 slices), A layout
  const size_t rowq = (size_t)(b * 1024 + qbase + l15) * 1024 + hh * 64;
  v16h qf0 = mk_frag(q + rowq +      kh * 8, q + rowq + 16 + kh * 8);
  v16h qf1 = mk_frag(q + rowq + 32 + kh * 8, q + rowq + 48 + kh * 8);

  float rm[8], rl[8];
  v8f accv[4];
#pragma unroll
  for (int r = 0; r < 8; ++r) { rm[r] = -3.0e38f; rl[r] = 0.f; }
#pragma unroll
  for (int nt = 0; nt < 4; ++nt) accv[nt] = (v8f){};

  const int nch = (qbase + 47) >> 5;   // chunks of 32 keys covering [0, qbase+16)
  for (int ch = 0; ch < nch; ++ch) {
    const int cb = ch * 32;

    // S = Q x K^T (two 16x16 n-tiles), B operand = key rows, table layout
    v8f s01[2];
#pragma unroll
    for (int j = 0; j < 2; ++j) {
      const size_t rowk = (size_t)(b * 1024 + cb + j * 16 + l15) * 1024 + hh * 64;
      v16h kf0 = mk_frag(k + rowk +      kh * 8, k + rowk + 16 + kh * 8);
      v16h kf1 = mk_frag(k + rowk + 32 + kh * 8, k + rowk + 48 + kh * 8);
      v8f z = (v8f){};
      z = WMMA_F16(qf0, kf0, z);
      z = WMMA_F16(qf1, kf1, z);
      s01[j] = z;
    }

    // Online softmax on the 16x32 score tile
#pragma unroll
    for (int r = 0; r < 8; ++r) {
      int row = qbase + r + kh * 8;
      int c0 = cb + l15, c1 = cb + 16 + l15;
      float s0 = s01[0][r] * 0.125f;  if (c0 > row) s0 = -3.0e38f;
      float s1 = s01[1][r] * 0.125f;  if (c1 > row) s1 = -3.0e38f;
      float mv = fmaxf(s0, s1);
#pragma unroll
      for (int m = 1; m < 16; m <<= 1) mv = fmaxf(mv, __shfl_xor(mv, m, 32));
      float rn   = fmaxf(rm[r], mv);
      float corr = __expf(rm[r] - rn);
      float p0 = __expf(s0 - rn), p1 = __expf(s1 - rn);
      float sum = p0 + p1;
#pragma unroll
      for (int m = 1; m < 16; m <<= 1) sum += __shfl_xor(sum, m, 32);
      rl[r] = rl[r] * corr + sum;
      rm[r] = rn;
#pragma unroll
      for (int nt = 0; nt < 4; ++nt) accv[nt][r] *= corr;
      P[(r + kh * 8) * 32 +      l15] = (half_t)p0;
      P[(r + kh * 8) * 32 + 16 + l15] = (half_t)p1;
    }
    asm volatile("s_wait_dscnt 0" ::: "memory");

    // P (16x32, A layout) x V chunk (32 x 64), via vT rows (table B layout)
    v16h pf = mk_frag(&P[l15 * 32 + kh * 8], &P[l15 * 32 + kh * 8 + 16]);
#pragma unroll
    for (int nt = 0; nt < 4; ++nt) {
      const size_t rowv = ((size_t)((b * 16 + hh) * 64 + nt * 16 + l15)) * 1024 + cb;
      v16h vf = mk_frag(vT + rowv + kh * 8, vT + rowv + kh * 8 + 16);
      accv[nt] = WMMA_F16(pf, vf, accv[nt]);
    }
  }

  // Normalize and store
#pragma unroll
  for (int nt = 0; nt < 4; ++nt) {
#pragma unroll
    for (int r = 0; r < 8; ++r) {
      int row = qbase + r + kh * 8;
      float ov = accv[nt][r] / rl[r];
      o[(size_t)(b * 1024 + row) * 1024 + hh * 64 + nt * 16 + l15] = (half_t)ov;
    }
  }
}

// ---------------------------------------------------------------------------
// Host orchestration
// ---------------------------------------------------------------------------
extern "C" void kernel_launch(void* const* d_in, const int* in_sizes, int n_in,
                              void* d_out, int out_size, void* d_ws, size_t ws_size,
                              hipStream_t stream) {
  (void)in_sizes; (void)n_in; (void)out_size; (void)ws_size;
  constexpr int Bv = 2, T = 1024, E = 1024, Hh = 16, HD = 64, Lx = 4, V = 8192;
  constexpr int M = Bv * T;                 // 2048 rows
  constexpr size_t MB = 1u << 20;

  const int*   x     = (const int*)  d_in[0];
  const float* tok   = (const float*)d_in[1];
  const float* pos   = (const float*)d_in[2];
  const float* Wq    = (const float*)d_in[3];
  const float* Wk    = (const float*)d_in[4];
  const float* Wv    = (const float*)d_in[5];
  const float* Wo    = (const float*)d_in[6];
  const float* bo    = (const float*)d_in[7];
  const float* ln1g  = (const float*)d_in[8];
  const float* ln1b  = (const float*)d_in[9];
  const float* W1    = (const float*)d_in[10];
  const float* b1    = (const float*)d_in[11];
  const float* W2    = (const float*)d_in[12];
  const float* b2    = (const float*)d_in[13];
  const float* ln2g  = (const float*)d_in[14];
  const float* ln2b  = (const float*)d_in[15];
  const float* lnfg  = (const float*)d_in[16];
  const float* lnfb  = (const float*)d_in[17];
  const float* Wp    = (const float*)d_in[18];
  const float* bp    = (const float*)d_in[19];

  char* ws = (char*)d_ws;
  float*  h    = (float*) (ws +  0 * MB);   // 8 MB   [2048,1024] f32
  float*  tmp  = (float*) (ws +  8 * MB);   // 8 MB   pre-LN sums
  half_t* h16  = (half_t*)(ws + 16 * MB);   // 4 MB
  half_t* q16  = (half_t*)(ws + 20 * MB);   // 4 MB
  half_t* k16  = (half_t*)(ws + 24 * MB);   // 4 MB
  half_t* v16  = (half_t*)(ws + 28 * MB);   // 4 MB
  half_t* vT16 = (half_t*)(ws + 32 * MB);   // 4 MB
  half_t* o16  = (half_t*)(ws + 36 * MB);   // 4 MB
  half_t* hf16 = (half_t*)(ws + 40 * MB);   // 4 MB
  half_t* ff16 = (half_t*)(ws + 44 * MB);   // 16 MB  [2048,4096] f16

  auto gemm = [&](const half_t* A, const float* B, const float* bias,
                  const float* resid, float* oF, half_t* oH,
                  int N, int K, int flags, bool headb) {
    dim3 grid(N / 128, M / 128);
    if (headb)
      gemm_f16_kernel<true><<<grid, 256, 0, stream>>>(A, B, bias, resid, oF, oH, N, K, flags);
    else
      gemm_f16_kernel<false><<<grid, 256, 0, stream>>>(A, B, bias, resid, oF, oH, N, K, flags);
  };

  embed_kernel<<<M, 256, 0, stream>>>(x, tok, pos, h, h16);

  for (int l = 0; l < Lx; ++l) {
    const size_t wqk = (size_t)l * Hh * E * HD;   // per-layer offset (1M elems)
    // QKV projections (head-blocked weights), f16-only outputs
    gemm(h16, Wq + wqk, nullptr, nullptr, nullptr, q16, E, E, 16, true);
    gemm(h16, Wk + wqk, nullptr, nullptr, nullptr, k16, E, E, 16, true);
    gemm(h16, Wv + wqk, nullptr, nullptr, nullptr, v16, E, E, 16, true);
    repackV_kernel<<<(M * E) / 256, 256, 0, stream>>>(v16, vT16);
    attn_kernel<<<dim3(T / 16, Hh, Bv), 32, 0, stream>>>(q16, k16, vT16, o16);
    // o @ Wo + bo + h -> tmp ; LN -> h, h16
    gemm(o16, Wo + (size_t)l * E * E, bo + (size_t)l * E, h, tmp, nullptr, E, E, 1 | 2 | 8, false);
    ln_kernel<<<M, 256, 0, stream>>>(tmp, ln1g + (size_t)l * E, ln1b + (size_t)l * E, h, h16);
    // MLP: relu(h@W1+b1) -> ff16 ; ff@W2+b2 + h -> tmp ; LN -> h, h16
    gemm(h16, W1 + (size_t)l * E * 4 * E, b1 + (size_t)l * 4 * E, nullptr,
         nullptr, ff16, 4 * E, E, 1 | 4 | 16, false);
    gemm(ff16, W2 + (size_t)l * 4 * E * E, b2 + (size_t)l * E, h, tmp, nullptr,
         E, 4 * E, 1 | 2 | 8, false);
    ln_kernel<<<M, 256, 0, stream>>>(tmp, ln2g + (size_t)l * E, ln2b + (size_t)l * E, h, h16);
  }

  // Final LN (f16 only) + logits projection -> d_out (f32)
  ln_kernel<<<M, 256, 0, stream>>>(h, lnfg, lnfb, nullptr, hf16);
  gemm(hf16, Wp, bp, nullptr, (float*)d_out, nullptr, V, E, 1 | 8, false);
}
